// Net_84207128806020
// MI455X (gfx1250) — compile-verified, matrix operand order checked by reading, not run on hardware
//
#include <hip/hip_runtime.h>

typedef __attribute__((ext_vector_type(16))) _Float16 v16h;
typedef __attribute__((ext_vector_type(8)))  float    v8f;

// ---------------------------------------------------------------------------
// Fragment load for V_WMMA_F32_16X16X32_F16.
// A-matrix ISA layout (16-bit, 16x32): lanes 0-15 hold M=lane; lanes 16-31 the
// same rows with the other K half:  dword p of the v16h holds K pair starting
// at  k0 + 8*(lane>=16) + 16*(p>=4) + 2*(p&3).
// B is stored transposed row-major [N][K], so the identical gather produces the
// mirrored B layout (lane -> column N).  Dwords 0-3 and 4-7 are contiguous, so
// the compiler lowers each fragment to two ds_load_b128 / global_load_b128.
// ---------------------------------------------------------------------------
__device__ __forceinline__ v16h load_frag(const _Float16* __restrict__ base,
                                          int ldk, int k0, int lane) {
  union { v16h v; unsigned int d[8]; } u;
  const int row = lane & 15;
  const _Float16* rp = base + row * ldk + k0 + ((lane >> 4) & 1) * 8;
#pragma unroll
  for (int p = 0; p < 8; ++p) {
    u.d[p] = *(const unsigned int*)(rp + ((p >> 2) * 16 + ((p & 3) * 2)));
  }
  return u.v;
}

__device__ __forceinline__ v8f wmma_f16(v16h a, v16h b, v8f c) {
  // (neg_a, A, neg_b, B, c_mod, C, reuse_a, reuse_b)
  return __builtin_amdgcn_wmma_f32_16x16x32_f16(false, a, false, b, (short)0, c,
                                                false, false);
}

// In-order DS pipe + this wait makes intra-wave LDS RAW safe; "memory" clobber
// stops compiler reordering around the fill->fragment-read boundary.
__device__ __forceinline__ void lds_fence() {
  asm volatile("s_wait_dscnt 0" ::: "memory");
}

// Async global->LDS copies (CDNA5 Tensor/Async path, tracked by ASYNCcnt).
// LDS destination address = low 32 bits of the generic pointer (workgroup-
// relative LDS offset).
__device__ __forceinline__ unsigned lds_addr(const void* p) {
  return (unsigned)(unsigned long long)p;
}
__device__ __forceinline__ void async_copy_b128(unsigned lds, const void* g) {
  asm volatile("global_load_async_to_lds_b128 %0, %1, off"
               :: "v"(lds), "v"(g) : "memory");
}
__device__ __forceinline__ void async_copy_b64(unsigned lds, const void* g) {
  asm volatile("global_load_async_to_lds_b64 %0, %1, off"
               :: "v"(lds), "v"(g) : "memory");
}
__device__ __forceinline__ void async_copy_b32(unsigned lds, const void* g) {
  asm volatile("global_load_async_to_lds_b32 %0, %1, off"
               :: "v"(lds), "v"(g) : "memory");
}
__device__ __forceinline__ void async_wait0() {
  asm volatile("s_wait_asynccnt 0" ::: "memory");
}

// act_quant(max(v,0), s, 0, qmax): relu -> /s -> clip -> rint(RNE) -> *s
__device__ __forceinline__ float qact(float v, float s, float qmax) {
  float y = fmaxf(v, 0.0f) / s;
  y = fminf(y, qmax);
  return rintf(y) * s;
}

// ---------------------------------------------------------------------------
// Kernel 0: per-tensor weight quantization.  block 0..4 -> w1,w2,wf1,wf2,wf3.
// Writes integer quant levels (exact in f16) into zero-padded [Npad][Kpad]
// tiles (B^T layout for the GEMMs) and the float scale into scales[].
// ---------------------------------------------------------------------------
__global__ __launch_bounds__(256) void prep_weights_kernel(
    const float* __restrict__ w1, const float* __restrict__ w2,
    const float* __restrict__ wf1, const float* __restrict__ wf2,
    const float* __restrict__ wf3, float* __restrict__ scales,
    _Float16* __restrict__ wq1, _Float16* __restrict__ wq2,
    _Float16* __restrict__ wf1q, _Float16* __restrict__ wf2q,
    _Float16* __restrict__ wf3q) {
  __shared__ float red[256];
  __shared__ float s_sh;
  const int t = threadIdx.x;
  const float* src = nullptr;
  _Float16* dst = nullptr;
  int N = 0, K = 0, Np = 0, Kp = 0, sidx = 0;
  float qmax = 1.f;
  bool i8 = false;
  switch (blockIdx.x) {
    case 0: src = w1;  dst = wq1;  N = 6;   K = 75;  Np = 16;  Kp = 96;  qmax = 1.f;   i8 = false; sidx = 0; break;
    case 1: src = w2;  dst = wq2;  N = 16;  K = 150; Np = 16;  Kp = 160; qmax = 1.f;   i8 = false; sidx = 1; break;
    case 2: src = wf1; dst = wf1q; N = 100; K = 400; Np = 112; Kp = 416; qmax = 127.f; i8 = true;  sidx = 2; break;
    case 3: src = wf2; dst = wf2q; N = 50;  K = 100; Np = 64;  Kp = 128; qmax = 127.f; i8 = true;  sidx = 3; break;
    case 4: src = wf3; dst = wf3q; N = 10;  K = 50;  Np = 16;  Kp = 64;  qmax = 127.f; i8 = true;  sidx = 4; break;
    default: return;
  }
  const int nElem = N * K;
  float m = 0.f;
  for (int i = t; i < nElem; i += 256) m = fmaxf(m, fabsf(src[i]));
  red[t] = m;
  __syncthreads();
  for (int s = 128; s > 0; s >>= 1) {
    if (t < s) red[t] = fmaxf(red[t], red[t + s]);
    __syncthreads();
  }
  if (t == 0) {
    float sc = i8 ? (red[0] / 127.f + 1e-8f) : (red[0] + 1e-8f);
    scales[sidx] = sc;
    s_sh = sc;
  }
  __syncthreads();
  const float sc = s_sh;
  for (int i = t; i < Np * Kp; i += 256) {
    int n = i / Kp, k = i - n * Kp;
    float lv = 0.f;
    if (n < N && k < K) {
      float y = src[n * K + k] / sc;
      y = fminf(fmaxf(y, -qmax), qmax);
      lv = rintf(y);
    }
    dst[i] = (_Float16)lv;
  }
}

// ---------------------------------------------------------------------------
// Kernel 1: quant(x) -> conv1 (implicit GEMM, M=16 positions, K=75->96, N=6)
//           -> maxpool2 -> BN -> quantReLU.  One block per image.
// (Staging here is a transform, f32 -> quantize -> f16, so it must go through
//  VGPRs; no async copy.)
// ---------------------------------------------------------------------------
__global__ __launch_bounds__(256) void conv1_kernel(
    const float* __restrict__ x, const float* __restrict__ s_in_p,
    const float* __restrict__ b1, const float* __restrict__ bn1_s,
    const float* __restrict__ bn1_b, const float* __restrict__ s_r1_p,
    const float* __restrict__ scales, const _Float16* __restrict__ wq1,
    _Float16* __restrict__ h1) {
  __shared__ __attribute__((aligned(16))) _Float16 xq[3 * 32 * 32];
  __shared__ __attribute__((aligned(16))) float cout[6 * 28 * 28];
  __shared__ __attribute__((aligned(16))) _Float16 atile[8][16 * 96];
  const int t = threadIdx.x, wave = t >> 5, lane = t & 31;
  const int b = blockIdx.x;
  const float s_in = s_in_p[0];
  const float* xin = x + (size_t)b * 3072;
  for (int i = t; i < 3072; i += 256) {
    float v = xin[i] / s_in;
    v = fminf(fmaxf(v, -2.f), 1.f);
    xq[i] = (_Float16)(rintf(v) * s_in);
  }
  __syncthreads();
  const float s1 = scales[0];
  const v16h bw0 = load_frag(wq1, 96, 0, lane);
  const v16h bw1 = load_frag(wq1, 96, 32, lane);
  const v16h bw2 = load_frag(wq1, 96, 64, lane);
  const int ch = lane & 15;
  const float bias = (ch < 6) ? b1[ch] : 0.f;
  for (int mt = wave; mt < 49; mt += 8) {   // 49*16 = 784 = 28*28 positions
    _Float16* at = atile[wave];
    for (int i = lane; i < 16 * 96; i += 32) {
      int row = i / 96, k = i - row * 96;
      float v = 0.f;
      if (k < 75) {
        int p = mt * 16 + row;
        int oy = p / 28, ox = p - oy * 28;
        int c = k / 25, r = (k - c * 25) / 5, s = k - c * 25 - r * 5;
        v = (float)xq[c * 1024 + (oy + r) * 32 + (ox + s)];
      }
      at[i] = (_Float16)v;
    }
    lds_fence();
    v8f acc = {0.f, 0.f, 0.f, 0.f, 0.f, 0.f, 0.f, 0.f};
    acc = wmma_f16(load_frag(at, 96, 0, lane), bw0, acc);
    acc = wmma_f16(load_frag(at, 96, 32, lane), bw1, acc);
    acc = wmma_f16(load_frag(at, 96, 64, lane), bw2, acc);
    if (ch < 6) {
      const int mbase = (lane >> 4) * 8;
#pragma unroll
      for (int v = 0; v < 8; v++) {
        int pos = mt * 16 + mbase + v;
        cout[ch * 784 + pos] = acc[v] * s1 + bias;
      }
    }
    lds_fence();
  }
  __syncthreads();
  const float sr1 = s_r1_p[0];
  _Float16* h1o = h1 + (size_t)b * 1176;
  for (int i = t; i < 1176; i += 256) {  // 6 * 14 * 14
    int c = i / 196, rem = i - c * 196, y = rem / 14, xo = rem - y * 14;
    const float* cb = &cout[c * 784 + (2 * y) * 28 + 2 * xo];
    float m = fmaxf(fmaxf(cb[0], cb[1]), fmaxf(cb[28], cb[29]));
    float v = m * bn1_s[c] + bn1_b[c];
    h1o[i] = (_Float16)qact(v, sr1, 3.f);
  }
}

// ---------------------------------------------------------------------------
// Kernel 2: conv2 (K=150->160, N=16) -> maxpool2 -> BN -> quantReLU -> flatten.
// One block per image.  h1 image staged with async global->LDS B128 copies
// (2352 B = 147 x 16 B).
// ---------------------------------------------------------------------------
__global__ __launch_bounds__(256) void conv2_kernel(
    const _Float16* __restrict__ h1, const float* __restrict__ b2,
    const float* __restrict__ bn2_s, const float* __restrict__ bn2_b,
    const float* __restrict__ s_r2_p, const float* __restrict__ scales,
    const _Float16* __restrict__ wq2, _Float16* __restrict__ h2) {
  __shared__ __attribute__((aligned(16))) _Float16 him[6 * 196];
  __shared__ __attribute__((aligned(16))) float cout[16 * 100];
  __shared__ __attribute__((aligned(16))) _Float16 atile[8][16 * 160];
  const int t = threadIdx.x, wave = t >> 5, lane = t & 31;
  const int b = blockIdx.x;
  const _Float16* hin = h1 + (size_t)b * 1176;
  if (t < 147) {
    async_copy_b128(lds_addr(&him[t * 8]), hin + t * 8);
  }
  async_wait0();
  __syncthreads();
  const float s2 = scales[1];
  v16h bw[5];
#pragma unroll
  for (int kc = 0; kc < 5; kc++) bw[kc] = load_frag(wq2, 160, kc * 32, lane);
  const int ch = lane & 15;
  const float bias = b2[ch];
  for (int mt = wave; mt < 7; mt += 8) {  // 100 positions -> 7 tiles (last partial)
    _Float16* at = atile[wave];
    for (int i = lane; i < 16 * 160; i += 32) {
      int row = i / 160, k = i - row * 160;
      float v = 0.f;
      int p = mt * 16 + row;
      if (p < 100 && k < 150) {
        int oy = p / 10, ox = p - oy * 10;
        int c = k / 25, r = (k - c * 25) / 5, s = k - c * 25 - r * 5;
        v = (float)him[c * 196 + (oy + r) * 14 + (ox + s)];
      }
      at[i] = (_Float16)v;
    }
    lds_fence();
    v8f acc = {0.f, 0.f, 0.f, 0.f, 0.f, 0.f, 0.f, 0.f};
#pragma unroll
    for (int kc = 0; kc < 5; kc++)
      acc = wmma_f16(load_frag(at, 160, kc * 32, lane), bw[kc], acc);
    const int mbase = (lane >> 4) * 8;
#pragma unroll
    for (int v = 0; v < 8; v++) {
      int pos = mt * 16 + mbase + v;
      if (pos < 100) cout[ch * 100 + pos] = acc[v] * s2 + bias;
    }
    lds_fence();
  }
  __syncthreads();
  const float sr2 = s_r2_p[0];
  _Float16* h2o = h2 + (size_t)b * 400;
  for (int i = t; i < 400; i += 256) {  // flatten order c*25 + y*5 + x
    int c = i / 25, rem = i - c * 25, y = rem / 5, xo = rem - y * 5;
    const float* cb = &cout[c * 100 + (2 * y) * 10 + 2 * xo];
    float m = fmaxf(fmaxf(cb[0], cb[1]), fmaxf(cb[10], cb[11]));
    float v = m * bn2_s[c] + bn2_b[c];
    h2o[i] = (_Float16)qact(v, sr2, 3.f);
  }
}

// ---------------------------------------------------------------------------
// Kernel 3: fc1  [B,400] x [400,100]^T  (K=400->416, N=100->112)
// A-tile staged with async B128 copies (16 rows x 50 chunks); K-tail zeroed
// with DS stores.
// ---------------------------------------------------------------------------
__global__ __launch_bounds__(256) void fc1_kernel(
    const _Float16* __restrict__ h2, const _Float16* __restrict__ wf1q,
    const float* __restrict__ bf1, const float* __restrict__ s_r3_p,
    const float* __restrict__ scales, _Float16* __restrict__ h3) {
  __shared__ __attribute__((aligned(16))) _Float16 aL[16 * 416];
  const int t = threadIdx.x, wave = t >> 5, lane = t & 31;
  const int b0 = blockIdx.x * 16;
  for (int i = t; i < 800; i += 256) {           // 16 rows * 50 b128 chunks
    int row = i / 50, c = i - row * 50;
    async_copy_b128(lds_addr(&aL[row * 416 + c * 8]),
                    h2 + (size_t)(b0 + row) * 400 + c * 8);
  }
  { int row = t >> 4, k = 400 + (t & 15); aL[row * 416 + k] = (_Float16)0.f; }
  async_wait0();
  __syncthreads();
  const float sw = scales[2];
  const float sr = s_r3_p[0];
  for (int nt = wave; nt < 7; nt += 8) {
    v8f acc = {0.f, 0.f, 0.f, 0.f, 0.f, 0.f, 0.f, 0.f};
    const _Float16* bb = wf1q + (size_t)nt * 16 * 416;
#pragma unroll
    for (int kc = 0; kc < 13; kc++)
      acc = wmma_f16(load_frag(aL, 416, kc * 32, lane),
                     load_frag(bb, 416, kc * 32, lane), acc);
    int n = nt * 16 + (lane & 15);
    if (n < 100) {
      float bias = bf1[n];
      int mbase = (lane >> 4) * 8;
#pragma unroll
      for (int v = 0; v < 8; v++) {
        float val = acc[v] * sw + bias;
        h3[(size_t)(b0 + mbase + v) * 100 + n] = (_Float16)qact(val, sr, 3.f);
      }
    }
  }
}

// ---------------------------------------------------------------------------
// Kernel 4: fc2  [B,100] x [100,50]^T  (K=100->128, N=50->64)
// Rows are 200 B -> async B64 copies (16 rows x 25 chunks).
// ---------------------------------------------------------------------------
__global__ __launch_bounds__(256) void fc2_kernel(
    const _Float16* __restrict__ h3, const _Float16* __restrict__ wf2q,
    const float* __restrict__ bf2, const float* __restrict__ s_r4_p,
    const float* __restrict__ scales, _Float16* __restrict__ h4) {
  __shared__ __attribute__((aligned(16))) _Float16 aL[16 * 128];
  const int t = threadIdx.x, wave = t >> 5, lane = t & 31;
  const int b0 = blockIdx.x * 16;
  for (int i = t; i < 400; i += 256) {           // 16 rows * 25 b64 chunks
    int row = i / 25, c = i - row * 25;
    async_copy_b64(lds_addr(&aL[row * 128 + c * 4]),
                   h3 + (size_t)(b0 + row) * 100 + c * 4);
  }
  for (int i = t; i < 448; i += 256) {           // zero pad k=100..127
    int row = i / 28, k = 100 + (i - row * 28);
    aL[row * 128 + k] = (_Float16)0.f;
  }
  async_wait0();
  __syncthreads();
  const float sw = scales[3];
  const float sr = s_r4_p[0];
  for (int nt = wave; nt < 4; nt += 8) {
    v8f acc = {0.f, 0.f, 0.f, 0.f, 0.f, 0.f, 0.f, 0.f};
    const _Float16* bb = wf2q + (size_t)nt * 16 * 128;
#pragma unroll
    for (int kc = 0; kc < 4; kc++)
      acc = wmma_f16(load_frag(aL, 128, kc * 32, lane),
                     load_frag(bb, 128, kc * 32, lane), acc);
    int n = nt * 16 + (lane & 15);
    if (n < 50) {
      float bias = bf2[n];
      int mbase = (lane >> 4) * 8;
#pragma unroll
      for (int v = 0; v < 8; v++) {
        float val = acc[v] * sw + bias;
        h4[(size_t)(b0 + mbase + v) * 50 + n] = (_Float16)qact(val, sr, 3.f);
      }
    }
  }
}

// ---------------------------------------------------------------------------
// Kernel 5: fc3  [B,50] x [50,10]^T  (K=50->64, N=10->16), f32 out.
// One wave per 16-row tile; rows are 100 B -> async B32 copies.  Each wave
// waits only its own ASYNCcnt (no block barrier needed).
// ---------------------------------------------------------------------------
__global__ __launch_bounds__(256) void fc3_kernel(
    const _Float16* __restrict__ h4, const _Float16* __restrict__ wf3q,
    const float* __restrict__ bf3, const float* __restrict__ scales,
    float* __restrict__ out) {
  __shared__ __attribute__((aligned(16))) _Float16 aL[8][16 * 64];
  const int wave = threadIdx.x >> 5, lane = threadIdx.x & 31;
  const int b0 = (blockIdx.x * 8 + wave) * 16;
  _Float16* at = aL[wave];
  for (int i = lane; i < 400; i += 32) {         // 16 rows * 25 b32 chunks
    int row = i / 25, c = i - row * 25;
    async_copy_b32(lds_addr(&at[row * 64 + c * 2]),
                   h4 + (size_t)(b0 + row) * 50 + c * 2);
  }
  for (int i = lane; i < 224; i += 32) {         // zero pad k=50..63
    int row = i / 14, k = 50 + (i - row * 14);
    at[row * 64 + k] = (_Float16)0.f;
  }
  async_wait0();
  lds_fence();
  const float sw = scales[4];
  v8f acc = {0.f, 0.f, 0.f, 0.f, 0.f, 0.f, 0.f, 0.f};
  acc = wmma_f16(load_frag(at, 64, 0, lane), load_frag(wf3q, 64, 0, lane), acc);
  acc = wmma_f16(load_frag(at, 64, 32, lane), load_frag(wf3q, 64, 32, lane), acc);
  int n = lane & 15;
  if (n < 10) {
    float bias = bf3[n];
    int mbase = (lane >> 4) * 8;
#pragma unroll
    for (int v = 0; v < 8; v++)
      out[(size_t)(b0 + mbase + v) * 10 + n] = acc[v] * sw + bias;
  }
}

// ---------------------------------------------------------------------------
// Launcher
// ---------------------------------------------------------------------------
extern "C" void kernel_launch(void* const* d_in, const int* in_sizes, int n_in,
                              void* d_out, int out_size, void* d_ws, size_t ws_size,
                              hipStream_t stream) {
  (void)in_sizes; (void)n_in; (void)out_size; (void)ws_size;
  const float* x    = (const float*)d_in[0];
  const float* s_in = (const float*)d_in[1];
  const float* w1   = (const float*)d_in[2];
  const float* b1   = (const float*)d_in[3];
  const float* bn1s = (const float*)d_in[4];
  const float* bn1b = (const float*)d_in[5];
  const float* sr1  = (const float*)d_in[6];
  const float* w2   = (const float*)d_in[7];
  const float* b2   = (const float*)d_in[8];
  const float* bn2s = (const float*)d_in[9];
  const float* bn2b = (const float*)d_in[10];
  const float* sr2  = (const float*)d_in[11];
  const float* wf1  = (const float*)d_in[12];
  const float* bf1  = (const float*)d_in[13];
  const float* sr3  = (const float*)d_in[14];
  const float* wf2  = (const float*)d_in[15];
  const float* bf2  = (const float*)d_in[16];
  const float* sr4  = (const float*)d_in[17];
  const float* wf3  = (const float*)d_in[18];
  const float* bf3  = (const float*)d_in[19];

  char* ws = (char*)d_ws;
  float*    scales = (float*)   (ws + 0);
  _Float16* wq1    = (_Float16*)(ws + 256);        // 16*96*2   = 3072
  _Float16* wq2    = (_Float16*)(ws + 3584);       // 16*160*2  = 5120
  _Float16* wf1q   = (_Float16*)(ws + 8960);       // 112*416*2 = 93184
  _Float16* wf2q   = (_Float16*)(ws + 102400);     // 64*128*2  = 16384
  _Float16* wf3q   = (_Float16*)(ws + 118784);     // 16*64*2   = 2048
  _Float16* h1     = (_Float16*)(ws + 120832);     // 16384*1176*2 = 38535168
  _Float16* h2     = (_Float16*)(ws + 38656000ull);// 16384*400*2  = 13107200
  _Float16* h3     = (_Float16*)(ws + 51763200ull);// 16384*100*2  = 3276800
  _Float16* h4     = (_Float16*)(ws + 55040000ull);// 16384*50*2   = 1638400

  prep_weights_kernel<<<5, 256, 0, stream>>>(w1, w2, wf1, wf2, wf3, scales,
                                             wq1, wq2, wf1q, wf2q, wf3q);
  conv1_kernel<<<16384, 256, 0, stream>>>(x, s_in, b1, bn1s, bn1b, sr1, scales, wq1, h1);
  conv2_kernel<<<16384, 256, 0, stream>>>(h1, b2, bn2s, bn2b, sr2, scales, wq2, h2);
  fc1_kernel<<<1024, 256, 0, stream>>>(h2, wf1q, bf1, sr3, scales, h3);
  fc2_kernel<<<1024, 256, 0, stream>>>(h3, wf2q, bf2, sr4, scales, h4);
  fc3_kernel<<<128, 256, 0, stream>>>(h4, wf3q, bf3, scales, (float*)d_out);
}
